// RNN_72988674228583
// MI455X (gfx1250) — compile-verified, matrix operand order checked by reading, not run on hardware
//
#include <hip/hip_runtime.h>
#include <hip/hip_bf16.h>
#include <math.h>

// ---------------------------------------------------------------------------
// Leaky tanh RNN on gfx1250:
//   recurrence : v_wmma_f32_16x16x32_bf16, 3 independent accumulator chains
//                (W | Bs | Bu fused K-loop), unroll-4 to rotate operand regs
//                through the WMMA WAR hazard window (kills v_nop padding)
//   readout    : v_wmma_f32_16x16x4_f32, 4 interleaved accumulator chains
// ---------------------------------------------------------------------------

typedef __bf16 bf16_t;
typedef bf16_t v16bf __attribute__((ext_vector_type(16)));
typedef float  v8f   __attribute__((ext_vector_type(8)));
typedef float  v2f   __attribute__((ext_vector_type(2)));

#define DIM_H  1024
#define DIM_S  1024
#define DIM_U  1024
#define DIM_Z  128
#define BATCH  128
#define TSTEPS 512
#define ALPHA  0.1f

// --- fragment loaders (ISA 7.12.2: 16-bit A 16x32 layout) -------------------
// lane holds row (lane&15); K chunks [c0..c0+7] and [c0+16..c0+23], c0=8*(lane>>4)
__device__ __forceinline__ v16bf load_frag_bf16(const bf16_t* __restrict__ row, int kbase) {
    union { uint4 u; bf16_t h[8]; } lo, hi;
    lo.u = *reinterpret_cast<const uint4*>(row + kbase);
    hi.u = *reinterpret_cast<const uint4*>(row + kbase + 16);
    v16bf r;
#pragma unroll
    for (int i = 0; i < 8; ++i) { r[i] = lo.h[i]; r[8 + i] = hi.h[i]; }
    return r;
}

// fp32 source converted to bf16 on the fly (s_seq / u_seq tiles)
__device__ __forceinline__ v16bf load_frag_f32cvt(const float* __restrict__ row, int kbase) {
    union { float4 f; float s[4]; } a0, a1, b0, b1;
    a0.f = *reinterpret_cast<const float4*>(row + kbase);
    a1.f = *reinterpret_cast<const float4*>(row + kbase + 4);
    b0.f = *reinterpret_cast<const float4*>(row + kbase + 16);
    b1.f = *reinterpret_cast<const float4*>(row + kbase + 20);
    v16bf r;
#pragma unroll
    for (int i = 0; i < 4; ++i) {
        r[i]      = (bf16_t)a0.s[i];
        r[4 + i]  = (bf16_t)a1.s[i];
        r[8 + i]  = (bf16_t)b0.s[i];
        r[12 + i] = (bf16_t)b1.s[i];
    }
    return r;
}

// --- K0: convert weights to bf16, seed h state, copy h0 into output t=0 -----
__global__ void rnn_prep(const float* __restrict__ W, const float* __restrict__ Bs,
                         const float* __restrict__ Bu, const float* __restrict__ h0,
                         bf16_t* __restrict__ Wbf, bf16_t* __restrict__ Bsbf,
                         bf16_t* __restrict__ Bubf, bf16_t* __restrict__ hbf0,
                         float* __restrict__ h_out) {
    const int n = DIM_H * DIM_H;
    for (int i = blockIdx.x * blockDim.x + threadIdx.x; i < n; i += gridDim.x * blockDim.x) {
        Wbf[i]  = (bf16_t)W[i];
        Bsbf[i] = (bf16_t)Bs[i];
        Bubf[i] = (bf16_t)Bu[i];
        if (i < BATCH * DIM_H) {
            float v  = h0[i];
            hbf0[i]  = (bf16_t)v;
            int bidx = i >> 10, h = i & (DIM_H - 1);
            h_out[(size_t)bidx * TSTEPS * DIM_H + h] = v;  // t = 0 slot
        }
    }
}

// --- K1: one recurrence step. Fused K = [W | Bs | Bu], 3 acc chains ---------
__global__ void __launch_bounds__(256)
rnn_step(const float* __restrict__ s_seq, const float* __restrict__ u_seq,
         const float* __restrict__ bias,
         const bf16_t* __restrict__ Wbf, const bf16_t* __restrict__ Bsbf,
         const bf16_t* __restrict__ Bubf,
         const bf16_t* __restrict__ hbf_in, bf16_t* __restrict__ hbf_out,
         float* __restrict__ h_out, int t) {
    const int lane = threadIdx.x & 31;
    const int idx  = blockIdx.x * 8 + (threadIdx.x >> 5);  // 0..511 tiles
    const int m_t  = idx & 7;                              // batch tile (8)
    const int n_t  = idx >> 3;                             // hidden tile (64)
    const int rm   = m_t * 16 + (lane & 15);
    const int rn   = n_t * 16 + (lane & 15);
    const int c0   = (lane >> 4) * 8;

    const bf16_t* hrow  = hbf_in + rm * DIM_H;
    const bf16_t* wrow  = Wbf  + rn * DIM_H;   // B(k,n)=W[n][k] -> row n of W
    const bf16_t* bsrow = Bsbf + rn * DIM_S;
    const bf16_t* burow = Bubf + rn * DIM_U;
    const float*  srow  = s_seq + ((size_t)rm * TSTEPS + t) * DIM_S;
    const float*  urow  = u_seq + ((size_t)rm * TSTEPS + t) * DIM_U;

    v8f acc0 = {};   // h @ W^T
    v8f acc1 = {};   // s_t @ Bs^T
    v8f acc2 = {};   // u_t @ Bu^T
#pragma unroll 4
    for (int k = 0; k < DIM_H; k += 32) {
        v16bf ah = load_frag_bf16(hrow, k + c0);
        v16bf bw = load_frag_bf16(wrow, k + c0);
        v16bf as = load_frag_f32cvt(srow, k + c0);
        v16bf bs = load_frag_bf16(bsrow, k + c0);
        v16bf au = load_frag_f32cvt(urow, k + c0);
        v16bf bu = load_frag_bf16(burow, k + c0);
        acc0 = __builtin_amdgcn_wmma_f32_16x16x32_bf16(false, ah, false, bw,
                                                       (short)0, acc0, false, false);
        acc1 = __builtin_amdgcn_wmma_f32_16x16x32_bf16(false, as, false, bs,
                                                       (short)0, acc1, false, false);
        acc2 = __builtin_amdgcn_wmma_f32_16x16x32_bf16(false, au, false, bu,
                                                       (short)0, acc2, false, false);
    }

    // C layout: VGPR j -> M = j + 8*(lane>>4), N = lane&15
#pragma unroll
    for (int j = 0; j < 8; ++j) {
        int m = m_t * 16 + j + ((lane >> 4) << 3);
        int n = n_t * 16 + (lane & 15);
        float pre   = (acc0[j] + acc1[j]) + (acc2[j] + bias[n]);
        size_t base = ((size_t)m * TSTEPS + t) * DIM_H + n;
        float hp    = h_out[base];                                   // h_t (fp32 truth)
        float hn    = (1.0f - ALPHA) * hp + ALPHA * tanhf(pre);
        h_out[base + DIM_H]    = hn;                                 // h_{t+1}
        hbf_out[m * DIM_H + n] = (bf16_t)hn;                         // next A operand
    }
}

// --- K2: readout z = h_seq @ Wz^T + bz, exact fp32 WMMA, 4 acc chains --------
__global__ void __launch_bounds__(256)
rnn_readout(const float* __restrict__ h_seq, const float* __restrict__ Wz,
            const float* __restrict__ bz, float* __restrict__ z_out) {
    const int lane = threadIdx.x & 31;
    const int idx  = blockIdx.x * 8 + (threadIdx.x >> 5);  // 0..32767 tiles
    const int n_t  = idx & 7;                              // Z tiles (8)
    const int m_t  = idx >> 3;                             // B*T tiles (4096)
    const int rm   = m_t * 16 + (lane & 15);
    const int rn   = n_t * 16 + (lane & 15);
    const int kc   = (lane >> 4) * 2;                      // f32 A 16x4 layout

    const float* arow = h_seq + (size_t)rm * DIM_H;
    const float* brow = Wz + rn * DIM_H;                   // B(k,n)=Wz[n][k]

    v8f acc[4] = {{}, {}, {}, {}};
#pragma unroll 2
    for (int k = 0; k < DIM_H; k += 16) {
#pragma unroll
        for (int q = 0; q < 4; ++q) {
            v2f a, b;
            float2 av = *reinterpret_cast<const float2*>(arow + k + 4 * q + kc);
            float2 bv = *reinterpret_cast<const float2*>(brow + k + 4 * q + kc);
            a[0] = av.x; a[1] = av.y;
            b[0] = bv.x; b[1] = bv.y;
            acc[q] = __builtin_amdgcn_wmma_f32_16x16x4_f32(false, a, false, b,
                                                           (short)0, acc[q], false, false);
        }
    }
#pragma unroll
    for (int j = 0; j < 8; ++j) {
        int m = m_t * 16 + j + ((lane >> 4) << 3);
        int n = n_t * 16 + (lane & 15);
        float v = (acc[0][j] + acc[1][j]) + (acc[2][j] + acc[3][j]);
        z_out[(size_t)m * DIM_Z + n] = v + bz[n];
    }
}

// ---------------------------------------------------------------------------
extern "C" void kernel_launch(void* const* d_in, const int* in_sizes, int n_in,
                              void* d_out, int out_size, void* d_ws, size_t ws_size,
                              hipStream_t stream) {
    const float* h0    = (const float*)d_in[0];
    const float* s_seq = (const float*)d_in[1];
    const float* u_seq = (const float*)d_in[2];
    const float* W     = (const float*)d_in[3];
    const float* bias  = (const float*)d_in[4];
    const float* Bs    = (const float*)d_in[5];
    const float* Bu    = (const float*)d_in[6];
    const float* Wz    = (const float*)d_in[7];
    const float* bz    = (const float*)d_in[8];

    float* h_out = (float*)d_out;                                  // [B,T,H]
    float* z_out = h_out + (size_t)BATCH * TSTEPS * DIM_H;         // [B,T,Z]

    bf16_t* Wbf  = (bf16_t*)d_ws;                                  // 2 MB
    bf16_t* Bsbf = Wbf  + (size_t)DIM_H * DIM_H;                   // 2 MB
    bf16_t* Bubf = Bsbf + (size_t)DIM_H * DIM_S;                   // 2 MB
    bf16_t* hbf0 = Bubf + (size_t)DIM_H * DIM_U;                   // 256 KB
    bf16_t* hbf1 = hbf0 + (size_t)BATCH * DIM_H;                   // 256 KB
    bf16_t* hbf[2] = { hbf0, hbf1 };

    rnn_prep<<<512, 256, 0, stream>>>(W, Bs, Bu, h0, Wbf, Bsbf, Bubf, hbf[0], h_out);

    for (int t = 0; t < TSTEPS - 1; ++t) {
        rnn_step<<<64, 256, 0, stream>>>(s_seq, u_seq, bias, Wbf, Bsbf, Bubf,
                                         hbf[t & 1], hbf[(t + 1) & 1], h_out, t);
    }

    rnn_readout<<<4096, 256, 0, stream>>>(h_out, Wz, bz, z_out);
}
